// PointTransformerV1_2929167696547
// MI455X (gfx1250) — compile-verified, matrix operand order checked by reading, not run on hardware
//
#include <hip/hip_runtime.h>
#include <hip/hip_bf16.h>

// ---------------------------------------------------------------------------
// Types for CDNA5 WMMA (wave32): 16x16x32 bf16 -> f32 accumulate
// ---------------------------------------------------------------------------
typedef __attribute__((ext_vector_type(16))) __bf16          v16bf;
typedef __attribute__((ext_vector_type(8)))  unsigned short  us8;
typedef __attribute__((ext_vector_type(8)))  float           v8f;

__device__ __forceinline__ unsigned short f2bf(float f) {
  unsigned u = __builtin_bit_cast(unsigned, f);
  unsigned r = u + 0x7FFFu + ((u >> 16) & 1u);   // round-to-nearest-even
  return (unsigned short)(r >> 16);
}

__device__ __forceinline__ v16bf cat16(us8 lo, us8 hi) {
  return __builtin_bit_cast(v16bf, __builtin_shufflevector(
      lo, hi, 0, 1, 2, 3, 4, 5, 6, 7, 8, 9, 10, 11, 12, 13, 14, 15));
}

#define GF_RELU 1
#define GF_BN   2
#define GF_RES  4
#define GF_BIAS 8

// ---------------------------------------------------------------------------
// GEMM: C[M,N] = act(A[M,K] * W[K,N] (+bias) (*bn)) (+resid)
// A pre-cast to bf16 [M, Kpad] (zero-padded), W pre-cast+transposed to bf16
// [Npad64, Kpad] (zero-padded). Software-pipelined inner loop: loads for
// K-step k0+32 are issued before the 4 WMMAs of step k0 (double-buffered
// fragments), so vmem overlaps matrix math. One wave -> 16(M) x 64(N).
// 8 waves/block -> 128 x 64 block tile.
// ---------------------------------------------------------------------------
__global__ __launch_bounds__(256)
void k_gemm(const unsigned short* __restrict__ Ab,
            const unsigned short* __restrict__ Wt,
            const float* __restrict__ bias,
            const float* __restrict__ resid, int ldr,
            float* __restrict__ C, int ldc,
            int M, int Nout, int Kpad, int flags)
{
  const int lane = threadIdx.x & 31;
  const int wave = threadIdx.x >> 5;
  const int m0 = blockIdx.x * 128 + wave * 16;
  if (m0 >= M) return;                       // M is always a multiple of 16
  const int n0 = blockIdx.y * 64;
  const int kb = (lane >> 4) * 8;            // per-lane K base (ISA frag layout)

  const unsigned short* ap = Ab + (size_t)(m0 + (lane & 15)) * Kpad + kb;
  const unsigned short* bp = Wt + (size_t)(n0 + (lane & 15)) * Kpad + kb;
  const size_t bstride = (size_t)16 * Kpad;

  v8f acc[4] = {{}, {}, {}, {}};

  // ---- pipeline prologue: load first K tile ----
  us8 a0 = *(const us8*)(ap);
  us8 a1 = *(const us8*)(ap + 16);
  us8 b0[4], b1[4];
#pragma unroll
  for (int h = 0; h < 4; ++h) {
    b0[h] = *(const us8*)(bp + h * bstride);
    b1[h] = *(const us8*)(bp + h * bstride + 16);
  }

  // ---- steady state: issue loads for k0, compute WMMAs for k0-32 ----
  for (int k0 = 32; k0 < Kpad; k0 += 32) {
    __builtin_prefetch(ap + k0 + 32, 0, 1);  // speculative, tile after next
    us8 na0 = *(const us8*)(ap + k0);
    us8 na1 = *(const us8*)(ap + k0 + 16);
    us8 nb0[4], nb1[4];
#pragma unroll
    for (int h = 0; h < 4; ++h) {
      nb0[h] = *(const us8*)(bp + h * bstride + k0);
      nb1[h] = *(const us8*)(bp + h * bstride + k0 + 16);
    }
    v16bf a = cat16(a0, a1);
#pragma unroll
    for (int h = 0; h < 4; ++h) {
      acc[h] = __builtin_amdgcn_wmma_f32_16x16x32_bf16(
          false, a, false, cat16(b0[h], b1[h]), (short)0, acc[h], false, false);
    }
    a0 = na0; a1 = na1;
#pragma unroll
    for (int h = 0; h < 4; ++h) { b0[h] = nb0[h]; b1[h] = nb1[h]; }
  }

  // ---- pipeline drain: last K tile ----
  {
    v16bf a = cat16(a0, a1);
#pragma unroll
    for (int h = 0; h < 4; ++h) {
      acc[h] = __builtin_amdgcn_wmma_f32_16x16x32_bf16(
          false, a, false, cat16(b0[h], b1[h]), (short)0, acc[h], false, false);
    }
  }

  const float bnscale = 0.99999500003749978f;  // 1/sqrt(1+1e-5)
#pragma unroll
  for (int h = 0; h < 4; ++h) {
    int n = n0 + h * 16 + (lane & 15);
    if (n >= Nout) continue;
    v8f a = acc[h];
#pragma unroll
    for (int r = 0; r < 8; ++r) {
      int m = m0 + r + ((lane >> 4) << 3);   // C/D layout: lanes 16-31 hold M+8
      float v = a[r];
      if (flags & GF_BIAS) v += bias[n];
      if (flags & GF_BN)   v *= bnscale;
      if (flags & GF_RELU) v = v > 0.0f ? v : 0.0f;
      if (flags & GF_RES)  v += resid[(size_t)m * ldr + n];
      C[(size_t)m * ldc + n] = v;
    }
  }
}

// fp32 activations [M, Kd] (leading dim lda) -> bf16 [M, Kpad], zero-padded
__global__ void k_acast(const float* __restrict__ A, int lda, int M, int Kd, int Kpad,
                        unsigned short* __restrict__ out)
{
  int t = blockIdx.x * blockDim.x + threadIdx.x;
  if (t >= M * Kpad) return;
  int k = t % Kpad, m = t / Kpad;
  out[t] = (k < Kd) ? f2bf(A[(size_t)m * lda + k]) : (unsigned short)0;
}

// fp32 weights [Kd, Nout] -> bf16 transposed [Npad, Kpad], zero-padded
__global__ void k_wcast(const float* __restrict__ w, unsigned short* __restrict__ wt,
                        int Kd, int Nout, int Kpad, int Npad)
{
  int i = blockIdx.x * blockDim.x + threadIdx.x;
  if (i >= Npad * Kpad) return;
  int n = i / Kpad, k = i % Kpad;
  wt[i] = (n < Nout && k < Kd) ? f2bf(w[(size_t)k * Nout + n]) : (unsigned short)0;
}

// ---------------------------------------------------------------------------
// Brute-force KNN over 2-D coords; k<=16, sorted ascending (top_k smallest).
// ---------------------------------------------------------------------------
__global__ void k_knn(const float* __restrict__ qxyz, const float* __restrict__ sxyz,
                      int Bv, int M, int Nsrc, int kn,
                      int* __restrict__ oidx, float* __restrict__ odist)
{
  int t = blockIdx.x * blockDim.x + threadIdx.x;
  if (t >= Bv * M) return;
  int b = t / M, m = t % M;
  float qx = qxyz[((size_t)b * M + m) * 2 + 0];
  float qy = qxyz[((size_t)b * M + m) * 2 + 1];
  float bd[16]; int bi[16];
  for (int i = 0; i < kn; ++i) { bd[i] = 3.4e38f; bi[i] = 0; }
  const float* sp = sxyz + (size_t)b * Nsrc * 2;
  for (int j = 0; j < Nsrc; ++j) {
    float dx = qx - sp[2 * j], dy = qy - sp[2 * j + 1];
    float d = dx * dx + dy * dy;
    if (d < bd[kn - 1]) {
      int p = kn - 1;
      while (p > 0 && d < bd[p - 1]) { bd[p] = bd[p - 1]; bi[p] = bi[p - 1]; --p; }
      bd[p] = d; bi[p] = j;
    }
  }
  for (int i = 0; i < kn; ++i) {
    oidx[(size_t)t * kn + i] = bi[i];
    if (odist) odist[(size_t)t * kn + i] = bd[i];
  }
}

// ---------------------------------------------------------------------------
// Farthest point sampling. One block per batch, LDS argmax reduction.
// ---------------------------------------------------------------------------
__global__ __launch_bounds__(256)
void k_fps(const float* __restrict__ xyz, int Npts, int npoint, int* __restrict__ out)
{
  __shared__ float sval[256];
  __shared__ int   sidx[256];
  __shared__ float scx, scy;
  __shared__ int   sfar;
  const int b = blockIdx.x;
  const int tid = threadIdx.x;
  const float* xp = xyz + (size_t)b * Npts * 2;
  float dmin[16];
  const int per = (Npts + 255) / 256;
  for (int i = 0; i < per; ++i) dmin[i] = 1e10f;
  if (tid == 0) sfar = 0;
  __syncthreads();
  for (int it = 0; it < npoint; ++it) {
    if (tid == 0) {
      out[(size_t)b * npoint + it] = sfar;
      scx = xp[2 * sfar]; scy = xp[2 * sfar + 1];
    }
    __syncthreads();
    float bv = -1.0f; int bidx = 0;
    for (int i = 0; i < per; ++i) {
      int p = tid + i * 256;
      if (p < Npts) {
        float dx = xp[2 * p] - scx, dy = xp[2 * p + 1] - scy;
        float d = dx * dx + dy * dy;
        float dm = dmin[i]; dm = dm < d ? dm : d; dmin[i] = dm;
        if (dm > bv || (dm == bv && p < bidx)) { bv = dm; bidx = p; }
      }
    }
    sval[tid] = bv; sidx[tid] = bidx;
    __syncthreads();
    for (int s = 128; s > 0; s >>= 1) {
      if (tid < s) {
        float ov = sval[tid + s]; int oi = sidx[tid + s];
        if (ov > sval[tid] || (ov == sval[tid] && oi < sidx[tid])) { sval[tid] = ov; sidx[tid] = oi; }
      }
      __syncthreads();
    }
    if (tid == 0) sfar = sidx[0];
    __syncthreads();
  }
}

// ---------------------------------------------------------------------------
// Gather (index_points) with optional center subtraction, writing into a
// (possibly wider, offset) channel slot: out[b, mc, k, outOff + c].
// mode: 0 plain, 1 = val - center, 2 = center - val
// ---------------------------------------------------------------------------
__global__ void k_gather(const float* __restrict__ pts, int Np, int C,
                         const int* __restrict__ idx, int Mtot, int m0, int CH, int Kn,
                         const float* __restrict__ center, int mode,
                         float* __restrict__ out, int outC, int outOff, int Bv)
{
  int t = blockIdx.x * blockDim.x + threadIdx.x;
  int tot = Bv * CH * Kn * C;
  if (t >= tot) return;
  int c = t % C; int r = t / C;
  int k = r % Kn; r /= Kn;
  int mc = r % CH; int b = r / CH;
  int m = m0 + mc;
  int id = idx[((size_t)b * Mtot + m) * Kn + k];
  float v = pts[((size_t)b * Np + id) * C + c];
  if (mode) {
    float cv = center[((size_t)b * Mtot + m) * C + c];
    v = (mode == 1) ? (v - cv) : (cv - v);
  }
  out[(((size_t)b * CH + mc) * Kn + k) * (size_t)outC + outOff + c] = v;
}

// t = q - kk + pos over a query chunk, layout [B, CH, Kn, F]
__global__ void k_qkpos(const float* __restrict__ q, int Mtot, int m0,
                        const float* __restrict__ kk, const float* __restrict__ pos,
                        float* __restrict__ out, int CH, int Kn, int F, int Bv)
{
  int t = blockIdx.x * blockDim.x + threadIdx.x;
  int tot = Bv * CH * Kn * F;
  if (t >= tot) return;
  int f = t % F; int r = t / F;
  r /= Kn;                 // drop k
  int mc = r % CH; int b = r / CH;
  out[t] = q[((size_t)b * Mtot + m0 + mc) * F + f] - kk[t] + pos[t];
}

// softmax over neighbor axis (k) + weighted sum with (v + pos)
__global__ void k_attn(const float* __restrict__ attn, const float* __restrict__ v,
                       const float* __restrict__ pos, float* __restrict__ res,
                       int Mtot, int m0, int CH, int Kn, int F, int Bv, float scale)
{
  int t = blockIdx.x * blockDim.x + threadIdx.x;
  int tot = Bv * CH * F;
  if (t >= tot) return;
  int f = t % F; int r = t / F;
  int mc = r % CH; int b = r / CH;
  size_t base = (((size_t)b * CH + mc) * Kn) * F + f;
  float a[16];
  float mx = -3.4e38f;
  for (int k = 0; k < Kn; ++k) {
    a[k] = attn[base + (size_t)k * F] * scale;
    mx = a[k] > mx ? a[k] : mx;
  }
  float s = 0.0f;
  for (int k = 0; k < Kn; ++k) { a[k] = __expf(a[k] - mx); s += a[k]; }
  float inv = 1.0f / s;
  float o = 0.0f;
  for (int k = 0; k < Kn; ++k)
    o += (a[k] * inv) * (v[base + (size_t)k * F] + pos[base + (size_t)k * F]);
  res[((size_t)b * Mtot + m0 + mc) * F + f] = o;
}

// max over neighbor axis: [BS, Kn, C] -> [BS, C]
__global__ void k_maxk(const float* __restrict__ h, float* __restrict__ out,
                       int BS, int Kn, int C)
{
  int t = blockIdx.x * blockDim.x + threadIdx.x;
  if (t >= BS * C) return;
  int c = t % C; int r = t / C;
  size_t base = ((size_t)r * Kn) * C + c;
  float mx = h[base];
  for (int k = 1; k < Kn; ++k) {
    float v = h[base + (size_t)k * C];
    mx = v > mx ? v : mx;
  }
  out[t] = mx;
}

// inverse-distance interpolation (transition_up)
__global__ void k_interp(const float* __restrict__ f1, int N1,
                         const int* __restrict__ idx, const float* __restrict__ d3,
                         const float* __restrict__ f2, float* __restrict__ out,
                         int N2, int C, int Bv)
{
  int t = blockIdx.x * blockDim.x + threadIdx.x;
  if (t >= Bv * N2 * C) return;
  int c = t % C; int r = t / C;
  int n = r % N2; int b = r / N2;
  size_t ib = ((size_t)b * N2 + n) * 3;
  float w0 = 1.0f / (d3[ib + 0] + 1e-8f);
  float w1 = 1.0f / (d3[ib + 1] + 1e-8f);
  float w2 = 1.0f / (d3[ib + 2] + 1e-8f);
  float s = w0 + w1 + w2;
  float o = (w0 * f1[((size_t)b * N1 + idx[ib + 0]) * C + c] +
             w1 * f1[((size_t)b * N1 + idx[ib + 1]) * C + c] +
             w2 * f1[((size_t)b * N1 + idx[ib + 2]) * C + c]) / s;
  out[((size_t)b * N2 + n) * C + c] = o + f2[((size_t)b * N2 + n) * C + c];
}

// ---------------------------------------------------------------------------
// Host-side orchestration
// ---------------------------------------------------------------------------
static inline int ceil_div(int a, int b) { return (a + b - 1) / b; }
static inline int pad32(int k) { return (k + 31) & ~31; }
static inline int pad64(int n) { return (n + 63) & ~63; }

struct Ctx {
  hipStream_t stream;
  char* ws; size_t off; size_t cap;
  void* alloc(size_t bytes) {
    size_t a = (off + 255) & ~size_t(255);
    off = a + bytes;
    return ws + a;
  }
};

struct LinP { const float* w; const float* b; int din, dout; };
struct TB { LinP fc1, fc2, d1, d2, g1, g2, wq, wk, wv; };

struct PC { void* const* in; int cur; };
static LinP takeLin(PC& pc, int din, int dout, bool bias) {
  LinP L;
  L.w = (const float*)pc.in[pc.cur++];
  L.b = bias ? (const float*)pc.in[pc.cur++] : nullptr;
  L.din = din; L.dout = dout;
  return L;
}
static TB takeTB(PC& pc, int dp) {
  TB t;
  t.fc1 = takeLin(pc, dp, 512, true);
  t.fc2 = takeLin(pc, 512, dp, true);
  t.d1  = takeLin(pc, 2, 512, true);
  t.d2  = takeLin(pc, 512, 512, true);
  t.g1  = takeLin(pc, 512, 512, true);
  t.g2  = takeLin(pc, 512, 512, true);
  t.wq  = takeLin(pc, 512, 512, false);
  t.wk  = takeLin(pc, 512, 512, false);
  t.wv  = takeLin(pc, 512, 512, false);
  return t;
}

static unsigned short* wcast(Ctx& c, const float* w, int Kd, int Nout) {
  int Kp = pad32(Kd), Np = pad64(Nout);
  unsigned short* wt = (unsigned short*)c.alloc((size_t)Kp * Np * 2);
  k_wcast<<<ceil_div(Kp * Np, 256), 256, 0, c.stream>>>(w, wt, Kd, Nout, Kp, Np);
  return wt;
}

// A is fp32 [M, Kd] contiguous; cast buffer is transient (freed after enqueue,
// safe: the single stream serializes later writers of the region behind us).
static void gemm_wt(Ctx& c, const float* A, int M, int Kd, const unsigned short* wt,
                    const float* bias, int Nout, float* Cp, int flags,
                    const float* resid, int ldr) {
  int Kp = pad32(Kd);
  size_t mark = c.off;
  unsigned short* ab = (unsigned short*)c.alloc((size_t)M * Kp * 2);
  k_acast<<<ceil_div(M * Kp, 256), 256, 0, c.stream>>>(A, Kd, M, Kd, Kp, ab);
  if (bias) flags |= GF_BIAS;
  dim3 g(ceil_div(M, 128), ceil_div(Nout, 64));
  k_gemm<<<g, 256, 0, c.stream>>>(ab, wt, bias, resid, ldr ? ldr : Nout,
                                  Cp, Nout, M, Nout, Kp, flags);
  c.off = mark;
}

static void gemm(Ctx& c, const float* A, int M, const LinP& L, float* Cp, int flags,
                 const float* resid = nullptr, int ldr = 0) {
  size_t mark = c.off;
  unsigned short* wt = wcast(c, L.w, L.din, L.dout);
  gemm_wt(c, A, M, L.din, wt, L.b, L.dout, Cp, flags, resid, ldr);
  c.off = mark;
}

// Point Transformer block: points <- fc2(attn(points)) + points, in place.
static void transformer(Ctx& c, const TB& tb, const float* xyz, float* points,
                        int M, int dp) {
  const size_t mark = c.off;
  const int F = 512, Kn = 16, Bv = 2;
  const int BM = Bv * M;

  int* idx = (int*)c.alloc((size_t)BM * Kn * 4);
  k_knn<<<ceil_div(BM, 256), 256, 0, c.stream>>>(xyz, xyz, Bv, M, M, Kn, idx, nullptr);

  float* x   = (float*)c.alloc((size_t)BM * F * 4);
  float* q   = (float*)c.alloc((size_t)BM * F * 4);
  float* xk  = (float*)c.alloc((size_t)BM * F * 4);
  float* xv  = (float*)c.alloc((size_t)BM * F * 4);
  float* res = (float*)c.alloc((size_t)BM * F * 4);

  gemm(c, points, BM, tb.fc1, x, 0);
  gemm(c, x, BM, tb.wq, q, 0);
  gemm(c, x, BM, tb.wk, xk, 0);
  gemm(c, x, BM, tb.wv, xv, 0);

  // weights reused across the chunk loop: keep them resident
  unsigned short* wd1 = wcast(c, tb.d1.w, 2,   512);
  unsigned short* wd2 = wcast(c, tb.d2.w, 512, 512);
  unsigned short* wg1 = wcast(c, tb.g1.w, 512, 512);
  unsigned short* wg2 = wcast(c, tb.g2.w, 512, 512);

  const int CH = M < 128 ? M : 128;        // chunk of queries, keeps tiles in L2
  const size_t cb = (size_t)Bv * CH * Kn * F * 4;
  float* rel = (float*)c.alloc((size_t)Bv * CH * Kn * 2 * 4);
  float* hid = (float*)c.alloc(cb);
  float* pos = (float*)c.alloc(cb);
  float* kk  = (float*)c.alloc(cb);
  float* vv  = (float*)c.alloc(cb);
  float* tt  = (float*)c.alloc(cb);
  float* at  = (float*)c.alloc(cb);

  for (int m0 = 0; m0 < M; m0 += CH) {
    int ch = (M - m0) < CH ? (M - m0) : CH;
    int rows = Bv * ch * Kn;
    // rel = xyz_center - xyz_neighbor (mode 2)
    k_gather<<<ceil_div(rows * 2, 256), 256, 0, c.stream>>>(
        xyz, M, 2, idx, M, m0, ch, Kn, xyz, 2, rel, 2, 0, Bv);
    gemm_wt(c, rel, rows, 2,   wd1, tb.d1.b, 512, hid, GF_RELU, nullptr, 0);
    gemm_wt(c, hid, rows, 512, wd2, tb.d2.b, 512, pos, 0, nullptr, 0);
    k_gather<<<ceil_div(rows * F, 256), 256, 0, c.stream>>>(
        xk, M, F, idx, M, m0, ch, Kn, nullptr, 0, kk, F, 0, Bv);
    k_gather<<<ceil_div(rows * F, 256), 256, 0, c.stream>>>(
        xv, M, F, idx, M, m0, ch, Kn, nullptr, 0, vv, F, 0, Bv);
    k_qkpos<<<ceil_div(rows * F, 256), 256, 0, c.stream>>>(
        q, M, m0, kk, pos, tt, ch, Kn, F, Bv);
    gemm_wt(c, tt,  rows, 512, wg1, tb.g1.b, 512, hid, GF_RELU, nullptr, 0);
    gemm_wt(c, hid, rows, 512, wg2, tb.g2.b, 512, at, 0, nullptr, 0);
    k_attn<<<ceil_div(Bv * ch * F, 256), 256, 0, c.stream>>>(
        at, vv, pos, res, M, m0, ch, Kn, F, Bv, 0.044194173824159216f); // 1/sqrt(512)
  }

  // points = linear(res, fc2) + points (residual)
  gemm(c, res, BM, tb.fc2, points, GF_RES, points, dp);
  c.off = mark;
}

static void tdown(Ctx& c, const LinP& m1, const LinP& m2,
                  const float* xyz, const float* pts, int Nin, int chin,
                  int S, float* nxyz, float* npts) {
  const size_t mark = c.off;
  const int Bv = 2, Kn = 16;
  int* fpsIdx = (int*)c.alloc((size_t)Bv * S * 4);
  k_fps<<<Bv, 256, 0, c.stream>>>(xyz, Nin, S, fpsIdx);
  // new_xyz = xyz[fpsIdx]  (gather with Kn=1)
  k_gather<<<ceil_div(Bv * S * 2, 256), 256, 0, c.stream>>>(
      xyz, Nin, 2, fpsIdx, S, 0, S, 1, nullptr, 0, nxyz, 2, 0, Bv);
  int* idx = (int*)c.alloc((size_t)Bv * S * Kn * 4);
  k_knn<<<ceil_div(Bv * S, 256), 256, 0, c.stream>>>(nxyz, xyz, Bv, S, Nin, Kn, idx, nullptr);
  const int cin = chin + 2;
  float* hin = (float*)c.alloc((size_t)Bv * S * Kn * cin * 4);
  k_gather<<<ceil_div(Bv * S * Kn * 2, 256), 256, 0, c.stream>>>(
      xyz, Nin, 2, idx, S, 0, S, Kn, nxyz, 1, hin, cin, 0, Bv);   // grouped - center
  k_gather<<<ceil_div(Bv * S * Kn * chin, 256), 256, 0, c.stream>>>(
      pts, Nin, chin, idx, S, 0, S, Kn, nullptr, 0, hin, cin, 2, Bv);
  const int chout = m1.dout;
  float* h1 = (float*)c.alloc((size_t)Bv * S * Kn * chout * 4);
  float* h2 = (float*)c.alloc((size_t)Bv * S * Kn * chout * 4);
  gemm(c, hin, Bv * S * Kn, m1, h1, GF_RELU | GF_BN);
  gemm(c, h1,  Bv * S * Kn, m2, h2, GF_RELU | GF_BN);
  k_maxk<<<ceil_div(Bv * S * chout, 256), 256, 0, c.stream>>>(h2, npts, Bv * S, Kn, chout);
  c.off = mark;
}

static void tup(Ctx& c, const LinP& f1l, const LinP& f2l,
                const float* xyzC, const float* ptsC, int Nc,
                const float* xyzF, const float* ptsF, int Nf, int chF,
                float* outF) {
  const size_t mark = c.off;
  const int Bv = 2;
  float* fe1 = (float*)c.alloc((size_t)Bv * Nc * chF * 4);
  float* fe2 = (float*)c.alloc((size_t)Bv * Nf * chF * 4);
  gemm(c, ptsC, Bv * Nc, f1l, fe1, GF_RELU | GF_BN);
  gemm(c, ptsF, Bv * Nf, f2l, fe2, GF_RELU | GF_BN);
  int*   idx = (int*)c.alloc((size_t)Bv * Nf * 3 * 4);
  float* d3  = (float*)c.alloc((size_t)Bv * Nf * 3 * 4);
  k_knn<<<ceil_div(Bv * Nf, 256), 256, 0, c.stream>>>(xyzF, xyzC, Bv, Nf, Nc, 3, idx, d3);
  k_interp<<<ceil_div(Bv * Nf * chF, 256), 256, 0, c.stream>>>(
      fe1, Nc, idx, d3, fe2, outF, Nf, chF, Bv);
  c.off = mark;
}

extern "C" void kernel_launch(void* const* d_in, const int* in_sizes, int n_in,
                              void* d_out, int out_size, void* d_ws, size_t ws_size,
                              hipStream_t stream) {
  (void)in_sizes; (void)n_in; (void)out_size;
  Ctx c; c.stream = stream; c.ws = (char*)d_ws; c.off = 0; c.cap = ws_size;

  PC pc{d_in, 0};
  const float* x = (const float*)pc.in[pc.cur++];   // [B,N,3]
  const float* p = (const float*)pc.in[pc.cur++];   // [B,N,2]

  LinP fc1a = takeLin(pc, 3, 32, true);
  LinP fc1b = takeLin(pc, 32, 32, true);
  TB tf1 = takeTB(pc, 32);
  LinP tdm[4][2]; TB tfd[4];
  for (int i = 0; i < 4; ++i) {
    int ch = 32 << (i + 1);
    tdm[i][0] = takeLin(pc, ch / 2 + 2, ch, true);
    tdm[i][1] = takeLin(pc, ch, ch, true);
  }
  for (int i = 0; i < 4; ++i) tfd[i] = takeTB(pc, 32 << (i + 1));
  LinP tul[4][2]; TB tfu[4];
  for (int i = 0; i < 4; ++i) {
    int ch = 32 << (3 - i);
    tul[i][0] = takeLin(pc, ch * 2, ch, true);
    tul[i][1] = takeLin(pc, ch, ch, true);
  }
  for (int i = 0; i < 4; ++i) tfu[i] = takeTB(pc, 32 << (3 - i));
  LinP fc2l[3] = { takeLin(pc, 512, 512, true), takeLin(pc, 512, 512, true),
                   takeLin(pc, 512, 512, true) };
  TB tf2 = takeTB(pc, 512);
  LinP fc3l[3] = { takeLin(pc, 32, 64, true), takeLin(pc, 64, 64, true),
                   takeLin(pc, 64, 3, true) };

  const int Ns[5] = {4096, 1024, 256, 64, 16};
  const int Cs[5] = {32, 64, 128, 256, 512};
  const float* xyzL[5]; float* xyzS[5]; float* ptsL[5];
  xyzL[0] = p; xyzS[0] = nullptr;
  for (int i = 1; i < 5; ++i) { xyzS[i] = (float*)c.alloc((size_t)2 * Ns[i] * 2 * 4); xyzL[i] = xyzS[i]; }
  for (int i = 0; i < 5; ++i) ptsL[i] = (float*)c.alloc((size_t)2 * Ns[i] * Cs[i] * 4);

  // h = mlp2(x, fc1[0], fc1[1])
  {
    size_t mark = c.off;
    float* t0 = (float*)c.alloc((size_t)8192 * 32 * 4);
    gemm(c, x, 8192, fc1a, t0, GF_RELU);
    gemm(c, t0, 8192, fc1b, ptsL[0], 0);
    c.off = mark;
  }
  transformer(c, tf1, xyzL[0], ptsL[0], Ns[0], Cs[0]);

  // Down path
  for (int i = 0; i < 4; ++i) {
    tdown(c, tdm[i][0], tdm[i][1], xyzL[i], ptsL[i], Ns[i], Cs[i],
          Ns[i + 1], xyzS[i + 1], ptsL[i + 1]);
    transformer(c, tfd[i], xyzL[i + 1], ptsL[i + 1], Ns[i + 1], Cs[i + 1]);
  }

  // fc2 head (L-R-L-R-L) at coarsest level, then transformer2
  {
    size_t mark = c.off;
    float* t0 = (float*)c.alloc((size_t)32 * 512 * 4);
    float* t1 = (float*)c.alloc((size_t)32 * 512 * 4);
    gemm(c, ptsL[4], 32, fc2l[0], t0, GF_RELU);
    gemm(c, t0, 32, fc2l[1], t1, GF_RELU);
    gemm(c, t1, 32, fc2l[2], ptsL[4], 0);
    c.off = mark;
  }
  transformer(c, tf2, xyzL[4], ptsL[4], Ns[4], Cs[4]);

  // Up path
  {
    float* cur = ptsL[4]; int cl = 4;
    for (int i = 0; i < 4; ++i) {
      int fine = 3 - i;
      tup(c, tul[i][0], tul[i][1], xyzL[cl], cur, Ns[cl],
          xyzL[fine], ptsL[fine], Ns[fine], Cs[fine], ptsL[fine]);
      transformer(c, tfu[i], xyzL[fine], ptsL[fine], Ns[fine], Cs[fine]);
      cur = ptsL[fine]; cl = fine;
    }
  }

  // fc3 (L-R-L-R-L), final GEMM straight into d_out [B,N,3]
  {
    size_t mark = c.off;
    float* t0 = (float*)c.alloc((size_t)8192 * 64 * 4);
    float* t1 = (float*)c.alloc((size_t)8192 * 64 * 4);
    gemm(c, ptsL[0], 8192, fc3l[0], t0, GF_RELU);
    gemm(c, t0, 8192, fc3l[1], t1, GF_RELU);
    gemm(c, t1, 8192, fc3l[2], (float*)d_out, 0);
    c.off = mark;
  }
}
